// AttentionWithPairBias_78228534329665
// MI455X (gfx1250) — compile-verified, matrix operand order checked by reading, not run on hardware
//
#include <hip/hip_runtime.h>

// ---------------------------------------------------------------------------
// AttentionWithPairBias for MI455X (gfx1250, wave32, WMMA 16x16x32 bf16)
//
// Sizes: B=1, L=768, C_S=384, C_Z=128, H=8, HD=48 (padded to 64 for Q/K).
// Memory-dominant term: pair tensor (302 MB > 192 MB L2, read once) ->
// fused LN + Wb projection, single streaming read with NT temporal hints
// (~13 us HBM floor at 23.3 TB/s). All GEMMs via v_wmma_f32_16x16x32_bf16.
// bf16 packing in the streaming kernel uses v_perm_b32 (1 op per pair).
// ---------------------------------------------------------------------------

#define LQ  768
#define CS  384
#define CZ  128
#define NH  8
#define HD  48
#define HDP 64
#define LL  (LQ * LQ)
#define EPSV 1e-5f

typedef unsigned short u16;
typedef __attribute__((ext_vector_type(16))) __bf16     v16bf;
typedef __attribute__((ext_vector_type(8)))  float      v8f;
typedef __attribute__((ext_vector_type(4)))  unsigned   v4u;
typedef __attribute__((ext_vector_type(4)))  float      v4f;

union Frag  { v16bf v; v4u q[2]; unsigned d[8]; u16 u[16]; };
union CFrag { v8f   v; float f[8]; };

__device__ __forceinline__ u16 f2bf(float f) {
  unsigned u = __builtin_bit_cast(unsigned, f);
  u += 0x7fffu + ((u >> 16) & 1u);
  return (u16)(u >> 16);
}

// two floats -> packed bf16 dword: +0x8000 (round-nearest, ties away) then a
// single v_perm_b32 byte-select of the two high halves.
__device__ __forceinline__ unsigned pack_bf16(float lo, float hi) {
  unsigned ul = __builtin_bit_cast(unsigned, lo) + 0x8000u;
  unsigned uh = __builtin_bit_cast(unsigned, hi) + 0x8000u;
  return __builtin_amdgcn_perm(uh, ul, 0x07060302u);
}

__device__ __forceinline__ v8f wmma_bf16(const Frag& a, const Frag& b, v8f c) {
  return __builtin_amdgcn_wmma_f32_16x16x32_bf16(false, a.v, false, b.v,
                                                 (short)0, c, false, false);
}

// ---------------------------------------------------------------------------
// prep: fp32 weights -> bf16; Wb -> fragment-packed [4][32][16] bf16 table
// (lane-contiguous so pair_bias loads it with 2x b128 per fragment);
// zero Q/K pad buffers.
// ---------------------------------------------------------------------------
__global__ __launch_bounds__(256) void prep_kernel(
    const float* __restrict__ Wq, const float* __restrict__ Wk,
    const float* __restrict__ Wv, const float* __restrict__ Wg,
    const float* __restrict__ Wo, const float* __restrict__ Wb,
    u16* wq, u16* wk, u16* wv, u16* wg, u16* wo, u16* wbf,
    u16* qpad, u16* kpad) {
  int i = blockIdx.x * 256 + threadIdx.x;               // grid covers 393216
  if (i < CS * CS) {
    wq[i] = f2bf(Wq[i]); wk[i] = f2bf(Wk[i]); wv[i] = f2bf(Wv[i]);
    wg[i] = f2bf(Wg[i]); wo[i] = f2bf(Wo[i]);
  }
  if (i < 4 * 32 * 16) {                                // Wb fragment packing
    int t = i >> 9, rem = i & 511;
    int lane = rem >> 4, e = rem & 15;
    int g = lane >> 4, r16 = lane & 15;
    int z = t * 32 + g * 8 + ((e < 8) ? e : e + 8);     // K index (0..127)
    wbf[i] = (r16 < NH) ? f2bf(Wb[z * NH + r16]) : (u16)0;
  }
  if (i < NH * LQ * HDP) { qpad[i] = 0; kpad[i] = 0; }
}

// ---------------------------------------------------------------------------
// LayerNorm(single) -> bf16  (one 128-thread block per row of 384)
// ---------------------------------------------------------------------------
__global__ __launch_bounds__(128) void ln_single_kernel(
    const float* __restrict__ x, const float* __restrict__ gw,
    const float* __restrict__ bw, u16* __restrict__ s_bf) {
  int row = blockIdx.x;
  int tid = threadIdx.x, lane = tid & 31, wave = tid >> 5;
  const float* xr = x + (size_t)row * CS;
  float v0 = xr[tid], v1 = xr[tid + 128], v2 = xr[tid + 256];
  float s1 = v0 + v1 + v2;
  float s2 = v0 * v0 + v1 * v1 + v2 * v2;
  #pragma unroll
  for (int m = 16; m; m >>= 1) {
    s1 += __shfl_xor(s1, m, 32);
    s2 += __shfl_xor(s2, m, 32);
  }
  __shared__ float sm[8];
  if (lane == 0) { sm[wave] = s1; sm[4 + wave] = s2; }
  __syncthreads();
  s1 = sm[0] + sm[1] + sm[2] + sm[3];
  s2 = sm[4] + sm[5] + sm[6] + sm[7];
  float mu = s1 * (1.0f / CS);
  float var = s2 * (1.0f / CS) - mu * mu;
  float rs = rsqrtf(var + EPSV);
  float v[3] = {v0, v1, v2};
  #pragma unroll
  for (int t = 0; t < 3; ++t) {
    int i = tid + t * 128;
    s_bf[(size_t)row * CS + i] = f2bf((v[t] - mu) * rs * gw[i] + bw[i]);
  }
}

// ---------------------------------------------------------------------------
// Generic WMMA GEMM:  C16x16 tiles of  A[768x384] @ W[384x384]^T
// MODE 0: store bf16 scaled -> padded [h][i][64]     (Q, K)
// MODE 1: store bf16 transposed -> [h][d][i]         (V)
// MODE 2: sigmoid(acc + bias) -> f32 gate buffer
// MODE 3: out = res + gate * (acc + bias) -> f32 d_out
// ---------------------------------------------------------------------------
template <int MODE>
__global__ __launch_bounds__(256) void gemm_kernel(
    const u16* __restrict__ A, const u16* __restrict__ W, float scale,
    const float* __restrict__ bias, const float* __restrict__ res,
    const float* __restrict__ gate, u16* __restrict__ out16,
    float* __restrict__ outf) {
  int lane = threadIdx.x & 31, wave = threadIdx.x >> 5;
  int tile = blockIdx.x * 8 + wave;          // 48*24 = 1152 tiles / 144 blocks
  int tm = tile / 24, tn = tile % 24;
  int m0 = tm * 16, n0 = tn * 16;
  int g = lane >> 4, r16 = lane & 15;

  const u16* arow = A + (size_t)(m0 + r16) * CS;
  const u16* brow = W + (size_t)(n0 + r16) * CS;

  CFrag c;
  #pragma unroll
  for (int r = 0; r < 8; ++r) c.f[r] = 0.0f;

  #pragma unroll 4
  for (int k0 = 0; k0 < CS; k0 += 32) {
    int kb = k0 + g * 8;
    Frag a, b;
    a.q[0] = *(const v4u*)(arow + kb);
    a.q[1] = *(const v4u*)(arow + kb + 16);
    b.q[0] = *(const v4u*)(brow + kb);
    b.q[1] = *(const v4u*)(brow + kb + 16);
    c.v = wmma_bf16(a, b, c.v);
  }

  #pragma unroll
  for (int r = 0; r < 8; ++r) {
    int Mg = m0 + r + 8 * g;
    int Ng = n0 + r16;
    if (MODE == 0) {                 // Q/K: [h][i][64] bf16, pre-scaled
      int h = n0 / HD, d = (n0 % HD) + r16;
      out16[((size_t)h * LQ + Mg) * HDP + d] = f2bf(c.f[r] * scale);
    } else if (MODE == 1) {          // V transposed: [h][d][i] bf16
      int h = n0 / HD, d = (n0 % HD) + r16;
      out16[((size_t)h * HD + d) * LQ + Mg] = f2bf(c.f[r]);
    } else if (MODE == 2) {          // gate = sigmoid(acc + bg)
      float z = c.f[r] + bias[Ng];
      outf[(size_t)Mg * CS + Ng] = 1.0f / (1.0f + __expf(-z));
    } else {                         // final residual epilogue
      size_t idx = (size_t)Mg * CS + Ng;
      outf[idx] = res[idx] + gate[idx] * (c.f[r] + bias[Ng]);
    }
  }
}

// ---------------------------------------------------------------------------
// Fused LN(pair) + pair-bias projection:  bias[h][i][j] = LN(z_ij) . Wb[:,h]
// One wave = 16 flat (i,j) rows; lanes l and l+16 split the 128-wide row.
// Pair loads are non-temporal (302 MB read-once stream, > L2 capacity).
// B-fragments from the pre-packed table: 2x b128 loads per fragment.
// 4x v_wmma_f32_16x16x32_bf16. Streams the pair tensor exactly once.
// ---------------------------------------------------------------------------
__global__ __launch_bounds__(256) void pair_bias_kernel(
    const float* __restrict__ pair, const float* __restrict__ gz,
    const float* __restrict__ bz, const u16* __restrict__ wbf,
    float* __restrict__ bias_out) {
  int lane = threadIdx.x & 31, wave = threadIdx.x >> 5;
  int g = lane >> 4, r16 = lane & 15;
  size_t m0 = (size_t)(blockIdx.x * 8 + wave) * 16;     // flat (i,j) tile base

  __shared__ float s_gz[CZ], s_bz[CZ];
  if (threadIdx.x < CZ) { s_gz[threadIdx.x] = gz[threadIdx.x];
                          s_bz[threadIdx.x] = bz[threadIdx.x]; }
  __syncthreads();

  // loop-invariant B fragments: contiguous per-lane b128 loads
  Frag b[4];
  #pragma unroll
  for (int t = 0; t < 4; ++t) {
    const v4u* p = (const v4u*)(wbf + ((size_t)t * 32 + lane) * 16);
    b[t].q[0] = p[0];
    b[t].q[1] = p[1];
  }

  const float* prow = pair + (m0 + r16) * (size_t)CZ;

  float vals[64];
  float s1 = 0.0f, s2 = 0.0f;
  #pragma unroll
  for (int ch = 0; ch < 8; ++ch) {
    const v4f* p4 = (const v4f*)(prow + ch * 16 + g * 8);
    v4f x0 = __builtin_nontemporal_load(p4);
    v4f x1 = __builtin_nontemporal_load(p4 + 1);
    #pragma unroll
    for (int e = 0; e < 4; ++e) {
      vals[ch * 8 + e]     = x0[e];
      vals[ch * 8 + 4 + e] = x1[e];
      s1 += x0[e] + x1[e];
      s2 += x0[e] * x0[e] + x1[e] * x1[e];
    }
  }
  s1 += __shfl_xor(s1, 16, 32);                 // partner lane holds other half
  s2 += __shfl_xor(s2, 16, 32);
  float mu = s1 * (1.0f / CZ);
  float var = s2 * (1.0f / CZ) - mu * mu;
  float rs = rsqrtf(var + EPSV);

  Frag a[4];
  #pragma unroll
  for (int t = 0; t < 4; ++t) {
    #pragma unroll
    for (int p = 0; p < 8; ++p) {               // one packed dword per step
      int ch = 2 * t + (p >> 2);
      int ee = (p & 3) * 2;
      int k = ch * 16 + g * 8 + ee;
      float y0 = (vals[ch * 8 + ee]     - mu) * rs * s_gz[k]     + s_bz[k];
      float y1 = (vals[ch * 8 + ee + 1] - mu) * rs * s_gz[k + 1] + s_bz[k + 1];
      a[t].d[p] = pack_bf16(y0, y1);
    }
  }

  CFrag c;
  #pragma unroll
  for (int r = 0; r < 8; ++r) c.f[r] = 0.0f;
  #pragma unroll
  for (int t = 0; t < 4; ++t) c.v = wmma_bf16(a[t], b[t], c.v);

  if (r16 < NH) {
    #pragma unroll
    for (int r = 0; r < 8; ++r)
      bias_out[(size_t)r16 * LL + m0 + r + 8 * g] = c.f[r];
  }
}

// ---------------------------------------------------------------------------
// Flash-style attention: one wave per (head, 16-query tile), online softmax
// over 24 chunks of 32 keys.  S via 4 WMMA, P->LDS layout bounce, PV via 3.
// Bias is a 19 MB read-once stream -> non-temporal loads.
// ---------------------------------------------------------------------------
__global__ __launch_bounds__(128) void attn_kernel(
    const u16* __restrict__ qw, const u16* __restrict__ kw,
    const u16* __restrict__ vtw, const float* __restrict__ bias,
    u16* __restrict__ o_bf) {
  int lane = threadIdx.x & 31, wave = threadIdx.x >> 5;
  int unit = blockIdx.x * 4 + wave;                      // 384 units / 96 blocks
  int h = unit & 7, q0 = (unit >> 3) * 16;
  int g = lane >> 4, r16 = lane & 15;

  __shared__ u16 pbuf_all[4][16 * 32];
  u16* pbuf = pbuf_all[wave];

  const u16* qrow = qw + ((size_t)h * LQ + q0 + r16) * HDP;
  Frag aq0, aq1;
  aq0.q[0] = *(const v4u*)(qrow + g * 8);
  aq0.q[1] = *(const v4u*)(qrow + g * 8 + 16);
  aq1.q[0] = *(const v4u*)(qrow + 32 + g * 8);
  aq1.q[1] = *(const v4u*)(qrow + 32 + g * 8 + 16);

  CFrag o0, o1, o2;
  float rmax[8], rsum[8];
  #pragma unroll
  for (int r = 0; r < 8; ++r) {
    o0.f[r] = o1.f[r] = o2.f[r] = 0.0f;
    rmax[r] = -1e30f; rsum[r] = 0.0f;
  }

  const float* brow = bias + (size_t)h * LL;

  for (int j0 = 0; j0 < LQ; j0 += 32) {
    CFrag s0, s1;
    #pragma unroll
    for (int r = 0; r < 8; ++r) { s0.f[r] = 0.0f; s1.f[r] = 0.0f; }

    const u16* kr0 = kw + ((size_t)h * LQ + j0 + r16) * HDP;
    const u16* kr1 = kr0 + 16 * HDP;
    Frag b;
    b.q[0] = *(const v4u*)(kr0 + g * 8);
    b.q[1] = *(const v4u*)(kr0 + g * 8 + 16);
    s0.v = wmma_bf16(aq0, b, s0.v);
    b.q[0] = *(const v4u*)(kr0 + 32 + g * 8);
    b.q[1] = *(const v4u*)(kr0 + 32 + g * 8 + 16);
    s0.v = wmma_bf16(aq1, b, s0.v);
    b.q[0] = *(const v4u*)(kr1 + g * 8);
    b.q[1] = *(const v4u*)(kr1 + g * 8 + 16);
    s1.v = wmma_bf16(aq0, b, s1.v);
    b.q[0] = *(const v4u*)(kr1 + 32 + g * 8);
    b.q[1] = *(const v4u*)(kr1 + 32 + g * 8 + 16);
    s1.v = wmma_bf16(aq1, b, s1.v);

    // online softmax over this 32-key chunk
    #pragma unroll
    for (int r = 0; r < 8; ++r) {
      int Mg = q0 + r + 8 * g;
      float b0 = __builtin_nontemporal_load(&brow[(size_t)Mg * LQ + j0 + r16]);
      float b1 = __builtin_nontemporal_load(&brow[(size_t)Mg * LQ + j0 + 16 + r16]);
      float e0 = s0.f[r] + b0, e1 = s1.f[r] + b1;
      float mx = fmaxf(e0, e1);
      #pragma unroll
      for (int m = 8; m; m >>= 1) mx = fmaxf(mx, __shfl_xor(mx, m, 32));
      float nm = fmaxf(rmax[r], mx);
      float sc = __expf(rmax[r] - nm);
      float p0 = __expf(e0 - nm), p1 = __expf(e1 - nm);
      float ps = p0 + p1;
      #pragma unroll
      for (int m = 8; m; m >>= 1) ps += __shfl_xor(ps, m, 32);
      rsum[r] = rsum[r] * sc + ps;
      rmax[r] = nm;
      o0.f[r] *= sc; o1.f[r] *= sc; o2.f[r] *= sc;
      int Ml = r + 8 * g;                       // C-layout -> LDS row-major P
      pbuf[Ml * 32 + r16]      = f2bf(p0);
      pbuf[Ml * 32 + 16 + r16] = f2bf(p1);
    }

    // reload P in A-fragment layout (per-wave LDS, DS ops are in-order)
    Frag ap;
    const u16* pl = pbuf + r16 * 32 + g * 8;
    ap.q[0] = *(const v4u*)(pl);
    ap.q[1] = *(const v4u*)(pl + 16);

    #pragma unroll
    for (int vt = 0; vt < 3; ++vt) {
      const u16* vrow = vtw + ((size_t)h * HD + vt * 16 + r16) * LQ + j0 + g * 8;
      Frag bv;
      bv.q[0] = *(const v4u*)(vrow);
      bv.q[1] = *(const v4u*)(vrow + 16);
      v8f acc = (vt == 0) ? o0.v : (vt == 1) ? o1.v : o2.v;
      acc = wmma_bf16(ap, bv, acc);
      if (vt == 0) o0.v = acc; else if (vt == 1) o1.v = acc; else o2.v = acc;
    }
  }

  #pragma unroll
  for (int r = 0; r < 8; ++r) {
    float inv = 1.0f / rsum[r];
    size_t Mg = q0 + r + 8 * g;
    o_bf[Mg * CS + h * HD + r16]      = f2bf(o0.f[r] * inv);
    o_bf[Mg * CS + h * HD + 16 + r16] = f2bf(o1.f[r] * inv);
    o_bf[Mg * CS + h * HD + 32 + r16] = f2bf(o2.f[r] * inv);
  }
}

// ---------------------------------------------------------------------------
// Workspace layout (bytes)
// ---------------------------------------------------------------------------
#define OFF_SBF   ((size_t)0)          // 589824   LN(single) bf16 [768x384]
#define OFF_GATE  ((size_t)589824)     // 1179648  gate f32 [768x384]
#define OFF_Q     ((size_t)1769472)    // 786432   Q bf16 [8][768][64]
#define OFF_K     ((size_t)2555904)    // 786432   K bf16 [8][768][64]
#define OFF_VT    ((size_t)3342336)    // 589824   V^T bf16 [8][48][768]
#define OFF_O     ((size_t)3932160)    // 589824   attn out bf16 [768x384]
#define OFF_WQ    ((size_t)4521984)    // 294912 x5 weights bf16
#define OFF_WK    ((size_t)4816896)
#define OFF_WV    ((size_t)5111808)
#define OFF_WG    ((size_t)5406720)
#define OFF_WO    ((size_t)5701632)
#define OFF_WB    ((size_t)5996544)    // 4096     Wb fragment table [4][32][16]
#define OFF_BIAS  ((size_t)6000640)    // 18874368 pair bias f32 [8][768][768]

extern "C" void kernel_launch(void* const* d_in, const int* in_sizes, int n_in,
                              void* d_out, int out_size, void* d_ws,
                              size_t ws_size, hipStream_t stream) {
  const float* single = (const float*)d_in[0];
  const float* pair   = (const float*)d_in[1];
  const float* g_s    = (const float*)d_in[2];
  const float* b_s    = (const float*)d_in[3];
  const float* g_z    = (const float*)d_in[4];
  const float* b_z    = (const float*)d_in[5];
  const float* Wq     = (const float*)d_in[6];
  const float* Wk     = (const float*)d_in[7];
  const float* Wv     = (const float*)d_in[8];
  const float* Wb     = (const float*)d_in[9];
  const float* Wo     = (const float*)d_in[10];
  const float* bo     = (const float*)d_in[11];
  const float* Wg     = (const float*)d_in[12];
  const float* bg     = (const float*)d_in[13];
  (void)in_sizes; (void)n_in; (void)out_size; (void)ws_size;

  char* ws = (char*)d_ws;
  u16*   s_bf  = (u16*)(ws + OFF_SBF);
  float* gate  = (float*)(ws + OFF_GATE);
  u16*   q_ws  = (u16*)(ws + OFF_Q);
  u16*   k_ws  = (u16*)(ws + OFF_K);
  u16*   vt_ws = (u16*)(ws + OFF_VT);
  u16*   o_ws  = (u16*)(ws + OFF_O);
  u16*   wq16  = (u16*)(ws + OFF_WQ);
  u16*   wk16  = (u16*)(ws + OFF_WK);
  u16*   wv16  = (u16*)(ws + OFF_WV);
  u16*   wg16  = (u16*)(ws + OFF_WG);
  u16*   wo16  = (u16*)(ws + OFF_WO);
  u16*   wbf   = (u16*)(ws + OFF_WB);
  float* biasb = (float*)(ws + OFF_BIAS);
  float* out   = (float*)d_out;

  prep_kernel<<<1536, 256, 0, stream>>>(Wq, Wk, Wv, Wg, Wo, Wb, wq16, wk16,
                                        wv16, wg16, wo16, wbf, q_ws, k_ws);
  ln_single_kernel<<<LQ, 128, 0, stream>>>(single, g_s, b_s, s_bf);

  const float qscale = 0.14433756729740643f;   // 1/sqrt(48)
  gemm_kernel<0><<<144, 256, 0, stream>>>(s_bf, wq16, qscale, nullptr, nullptr,
                                          nullptr, q_ws, nullptr);
  gemm_kernel<0><<<144, 256, 0, stream>>>(s_bf, wk16, 1.0f, nullptr, nullptr,
                                          nullptr, k_ws, nullptr);
  gemm_kernel<1><<<144, 256, 0, stream>>>(s_bf, wv16, 1.0f, nullptr, nullptr,
                                          nullptr, vt_ws, nullptr);
  gemm_kernel<2><<<144, 256, 0, stream>>>(s_bf, wg16, 1.0f, bg, nullptr,
                                          nullptr, nullptr, gate);

  pair_bias_kernel<<<4608, 256, 0, stream>>>(pair, g_z, b_z, wbf, biasb);

  attn_kernel<<<96, 128, 0, stream>>>(q_ws, k_ws, vt_ws, biasb, o_ws);

  gemm_kernel<3><<<144, 256, 0, stream>>>(o_ws, wo16, 1.0f, bo, single, gate,
                                          nullptr, out);
}
